// SparseUNetSpconv_7035156431609
// MI455X (gfx1250) — compile-verified, matrix operand order checked by reading, not run on hardware
//
#include <hip/hip_runtime.h>
#include <hip/hip_bf16.h>

// ---------------------------------------------------------------------------
// Sparse UNet (spconv-style, dense-with-masks) for gfx1250 / MI455X.
//
// Perf model: ~8 GFLOP total but ~280 MB of weights streamed once
//   -> memory bound on weight traffic (~12 us at 23.3 TB/s HBM).
// Implicit-GEMM convs on v_wmma_f32_16x16x32_f16 (f32 accumulate):
//   M = 16 output voxels, N = 16 output channels per wave, K = Cin*k^3
//   (K contiguous in OIDHW weights -> coalesced single-pass weight stream).
// Weight chunks are DMA'd to LDS with GLOBAL_LOAD_ASYNC_TO_LDS_B128 (ASYNCcnt)
// overlapping the A-tile gather; the K-loop is split into predicate-free full
// chunks (2x unrolled, double-buffered LDS) plus an optional tail. Single-wave
// workgroups: same-wave LDS ops are in-order, so no barriers are needed.
// ---------------------------------------------------------------------------

typedef __attribute__((ext_vector_type(16))) _Float16 v16h;
typedef __attribute__((ext_vector_type(8)))  _Float16 v8h;
typedef __attribute__((ext_vector_type(8)))  float    v8f;
typedef __attribute__((ext_vector_type(4)))  float    v4f;
typedef int v4i __attribute__((vector_size(16)));  // matches builtin prototype

#if defined(__HIP_DEVICE_COMPILE__) &&                                   \
    __has_builtin(__builtin_amdgcn_global_load_async_to_lds_b128) &&     \
    __has_builtin(__builtin_amdgcn_s_wait_asynccnt)
#define USE_ASYNC_LDS 1
#else
#define USE_ASYNC_LDS 0
#endif

struct CTrue  { static constexpr bool value = true;  };
struct CFalse { static constexpr bool value = false; };

#define EPSV 1e-5f
#define OCCV 0.1f

// ---------------- mask pyramid ----------------

__global__ void mask_init_kernel(const float* __restrict__ mask,
                                 float* __restrict__ m0, int n) {
  int i = blockIdx.x * blockDim.x + threadIdx.x;
  if (i < n) m0[i] = (mask[i] < OCCV) ? 1.0f : 0.0f;
}

// reduce_window(max, win=3, stride=2, pad=1): out[p] = max over in[2p-1 .. 2p+1]
__global__ void mask_down_kernel(const float* __restrict__ mi,
                                 float* __restrict__ mo, int Gin, int Gout) {
  int i = blockIdx.x * blockDim.x + threadIdx.x;
  int V = Gout * Gout * Gout;
  if (i >= V) return;
  int G2 = Gout * Gout;
  int z = i / G2, y = (i - z * G2) / Gout, x = i - z * G2 - y * Gout;
  float best = 0.0f;
  for (int dz = 0; dz < 3; ++dz) {
    int iz = 2 * z + dz - 1;
    if ((unsigned)iz >= (unsigned)Gin) continue;
    for (int dy = 0; dy < 3; ++dy) {
      int iy = 2 * y + dy - 1;
      if ((unsigned)iy >= (unsigned)Gin) continue;
      for (int dx = 0; dx < 3; ++dx) {
        int ix = 2 * x + dx - 1;
        if ((unsigned)ix >= (unsigned)Gin) continue;
        best = fmaxf(best, mi[(iz * Gin + iy) * Gin + ix]);
      }
    }
  }
  mo[i] = best;
}

__global__ void mulmask_kernel(const float* __restrict__ x,
                               const float* __restrict__ m,
                               float* __restrict__ y, int vmask, int n) {
  int i = blockIdx.x * blockDim.x + threadIdx.x;
  if (i < n) y[i] = x[i] * m[i & vmask];
}

// ---------------- WMMA implicit-GEMM conv ----------------
// MODE 0: forward conv, stride STRIDE, pad PAD, kernel 3 (KSPAN=27) or 1.
// MODE 1: stride-2 transposed conv, k=3, JAX 'SAME' -> pad (2,1) on dilated in:
//         out[p] = sum_k w[k] * x[(p+k-2)/2]  when (p+k-2) even & in range.

template <int CIN, int KSPAN, int MODE, int STRIDE, int PAD, int LGIN, int LGOUT>
__global__ __launch_bounds__(32)
void conv_wmma_kernel(const float* __restrict__ in, const float* __restrict__ w,
                      float* __restrict__ out) {
  constexpr int Gin = 1 << LGIN;
  constexpr int Gout = 1 << LGOUT;
  constexpr int Vout = 1 << (3 * LGOUT);
  constexpr int Ktot = CIN * KSPAN;
  constexpr int NFULL = Ktot / 32;            // predicate-free chunks
  constexpr bool HASTAIL = (Ktot % 32) != 0;  // only enc0 (Cin=8)

  __shared__ alignas(16) _Float16 lsA[2][16 * 32];  // activations, f16
  __shared__ alignas(16) float    lsB[2][16 * 32];  // weights, f32 (DMA target)

  const int t = threadIdx.x;
  const int m = t & 15;        // voxel row (A) / cout row (B) / N index (C)
  const int half = t >> 4;     // which 16-wide K half of the 32-K chunk
  const int vbase = blockIdx.x * 16;
  const int coBase = blockIdx.y * 16;

  // voxel coords for A gather: invariant across K chunks (hoisted)
  const int v = vbase + m;
  const bool vok = v < Vout;
  const int z = v >> (2 * LGOUT);
  const int y = (v >> LGOUT) & (Gout - 1);
  const int x = v & (Gout - 1);
  const int bz = (MODE == 0) ? (z * STRIDE - PAD) : (z - 2);
  const int by = (MODE == 0) ? (y * STRIDE - PAD) : (y - 2);
  const int bx = (MODE == 0) ? (x * STRIDE - PAD) : (x - 2);

  const float* __restrict__ wrow = w + (size_t)(coBase + m) * Ktot;

  v8f acc = {};

  auto chunk = [&](int c, auto fullc) {
    constexpr bool FULL = decltype(fullc)::value;
    const int kb = c * 32 + half * 16;  // this thread's 16-K slice
    _Float16* bufA = lsA[c & 1];
    float*    bufB = lsB[c & 1];

    bool asyncB = false;
#if USE_ASYNC_LDS
    if (FULL) {
      // DMA the 16x32 f32 weight chunk straight into LDS: 4 per-lane b128
      // async issues (2KB total), tracked on ASYNCcnt, overlapping the A
      // gather below.
#pragma unroll
      for (int j = 0; j < 4; ++j) {
        const int q = j * 128 + t * 4;        // flat f32 index in chunk
        const int row = q >> 5, col = q & 31; // row = cout, col = K
        const float* gp = w + (size_t)(coBase + row) * Ktot + c * 32 + col;
        __builtin_amdgcn_global_load_async_to_lds_b128(
            (__attribute__((address_space(1))) v4i*)gp,
            (__attribute__((address_space(3))) v4i*)&bufB[q], 0, 0);
      }
      asyncB = true;
    }
#endif
    if ((c + 1) * 32 < Ktot)  // pull next chunk toward L2 (global_prefetch)
      __builtin_prefetch(&wrow[kb + 32], 0, 1);

    // --- A gather (16 K values per thread) + sync B fallback ---
    v8h taL = {}, taH = {};
    v4f tb0 = {}, tb1 = {}, tb2 = {}, tb3 = {};
#pragma unroll
    for (int i = 0; i < 16; ++i) {
      const int kidx = kb + i;
      const bool kok = FULL || (kidx < Ktot);
      float av = 0.0f, bv = 0.0f;
      if (!asyncB && kok) bv = wrow[kidx];
      if (kok) {
        const int ci = kidx / KSPAN;           // constant divisor
        const int kr = kidx - ci * KSPAN;      // 0 when KSPAN==1
        const int kd = kr / 9;
        const int kh = (kr - kd * 9) / 3;
        const int kw = kr - kd * 9 - kh * 3;
        int iz = bz + kd, iy = by + kh, ix = bx + kw;
        bool ok;
        if (MODE == 0) {
          ok = (unsigned)iz < (unsigned)Gin && (unsigned)iy < (unsigned)Gin &&
               (unsigned)ix < (unsigned)Gin;
        } else {
          ok = iz >= 0 && iy >= 0 && ix >= 0 && !(iz & 1) && !(iy & 1) &&
               !(ix & 1);
          iz >>= 1; iy >>= 1; ix >>= 1;
          ok = ok && iz < Gin && iy < Gin && ix < Gin;
        }
        if (vok && ok)
          av = in[((size_t)ci << (3 * LGIN)) +
                  (((iz << LGIN) + iy) << LGIN) + ix];
      }
      const _Float16 ah = (_Float16)av;
      if (i < 8) taL[i] = ah; else taH[i - 8] = ah;
      if (i < 4)       tb0[i] = bv;
      else if (i < 8)  tb1[i - 4] = bv;
      else if (i < 12) tb2[i - 8] = bv;
      else             tb3[i - 12] = bv;
    }
    *(v8h*)&bufA[m * 32 + half * 16]     = taL;   // ds_store_b128 x2
    *(v8h*)&bufA[m * 32 + half * 16 + 8] = taH;
    if (!asyncB) {
      *(v4f*)&bufB[m * 32 + half * 16]      = tb0;
      *(v4f*)&bufB[m * 32 + half * 16 + 4]  = tb1;
      *(v4f*)&bufB[m * 32 + half * 16 + 8]  = tb2;
      *(v4f*)&bufB[m * 32 + half * 16 + 12] = tb3;
    }
#if USE_ASYNC_LDS
    if (asyncB) __builtin_amdgcn_s_wait_asynccnt(0);  // drain DMA into LDS
#endif

    // ISA 16-bit A layout: lanes 0-15 hold K 0-7 / 16-23; lanes 16-31 hold
    // K 8-15 / 24-31 -> two contiguous 16B runs per lane (ds_load_b128).
    const _Float16* pa = &bufA[m * 32 + (half ? 8 : 0)];
    const v8h A0 = *(const v8h*)pa;
    const v8h A1 = *(const v8h*)(pa + 16);
    const v16h afrag = __builtin_shufflevector(
        A0, A1, 0, 1, 2, 3, 4, 5, 6, 7, 8, 9, 10, 11, 12, 13, 14, 15);

    const float* pb = &bufB[m * 32 + (half ? 8 : 0)];
    const v4f B0 = *(const v4f*)pb;
    const v4f B1 = *(const v4f*)(pb + 4);
    const v4f B2 = *(const v4f*)(pb + 16);
    const v4f B3 = *(const v4f*)(pb + 20);
    v16h bfrag;
#pragma unroll
    for (int j = 0; j < 4; ++j) {
      bfrag[j]      = (_Float16)B0[j];
      bfrag[4 + j]  = (_Float16)B1[j];
      bfrag[8 + j]  = (_Float16)B2[j];
      bfrag[12 + j] = (_Float16)B3[j];
    }

    acc = __builtin_amdgcn_wmma_f32_16x16x32_f16(
        /*neg_a=*/false, afrag, /*neg_b=*/false, bfrag,
        /*c_mod=*/(short)0, acc, /*reuse_a=*/false, /*reuse_b=*/false);
  };

#pragma unroll 2
  for (int c = 0; c < NFULL; ++c) chunk(c, CTrue{});
  if (HASTAIL) chunk(NFULL, CFalse{});

  // C/D layout: VGPR r -> lanes 0-15: (M=r, N=lane); lanes 16-31: (M=r+8,
  // N=lane-16). Each lane owns 8 consecutive voxels of one channel column.
  const int co = coBase + m;
  const int mo = half * 8;
  float* op = out + (size_t)co * Vout + vbase + mo;
  if (vbase + mo + 8 <= Vout) {
    *(v4f*)op       = v4f{acc[0], acc[1], acc[2], acc[3]};
    *(v4f*)(op + 4) = v4f{acc[4], acc[5], acc[6], acc[7]};
  } else {
#pragma unroll
    for (int r = 0; r < 8; ++r) {
      int vv = vbase + mo + r;
      if (vv < Vout) out[(size_t)co * Vout + vv] = acc[r];
    }
  }
}

// ---------------- masked BN stats + BN/ReLU ----------------
// block c < C: st[c] = sum(x*m), st[C+c] = sum(x*x*m); block c == C: st[2C]=sum(m)

__global__ void bn_stats_kernel(const float* __restrict__ x,
                                const float* __restrict__ m,
                                float* __restrict__ st, int C, int V) {
  __shared__ float sA[256];
  __shared__ float sB[256];
  int c = blockIdx.x;
  int tid = threadIdx.x;
  float a = 0.0f, b = 0.0f;
  if (c < C) {
    const float* xc = x + (size_t)c * V;
    for (int v = tid; v < V; v += 256) {
      float xv = xc[v], mv = m[v];
      a += xv * mv;
      b += xv * xv * mv;
    }
  } else {
    for (int v = tid; v < V; v += 256) a += m[v];
  }
  sA[tid] = a; sB[tid] = b;
  __syncthreads();
  for (int s = 128; s > 0; s >>= 1) {
    if (tid < s) { sA[tid] += sA[tid + s]; sB[tid] += sB[tid + s]; }
    __syncthreads();
  }
  if (tid == 0) {
    if (c < C) { st[c] = sA[0]; st[C + c] = sB[0]; }
    else       { st[2 * C] = sA[0]; }
  }
}

__global__ void bn_relu_kernel(float* __restrict__ x,
                               const float* __restrict__ m,
                               const float* __restrict__ st, int C, int lgV) {
  int i = blockIdx.x * blockDim.x + threadIdx.x;
  int V = 1 << lgV;
  if (i >= (C << lgV)) return;
  int c = i >> lgV;
  int v = i & (V - 1);
  float n = fmaxf(st[2 * C], 1.0f);
  float mu = st[c] / n;
  float var = st[C + c] / n - mu * mu;
  float yv = (x[i] - mu) * rsqrtf(var + EPSV);
  x[i] = fmaxf(yv, 0.0f) * m[v];
}

// ---------------- host-side pipeline ----------------

template <int CIN, int KSPAN, int MODE, int STRIDE, int PAD, int LGIN, int LGOUT>
static inline void run_conv(const float* in, const float* w, float* out,
                            int Cout, hipStream_t s) {
  constexpr int Vout = 1 << (3 * LGOUT);
  dim3 grid((Vout + 15) / 16, Cout / 16);
  conv_wmma_kernel<CIN, KSPAN, MODE, STRIDE, PAD, LGIN, LGOUT>
      <<<grid, 32, 0, s>>>(in, w, out);
}

static inline void run_bn(float* x, const float* m, float* st, int C, int lgV,
                          hipStream_t s) {
  int V = 1 << lgV;
  bn_stats_kernel<<<C + 1, 256, 0, s>>>(x, m, st, C, V);
  int n = C << lgV;
  bn_relu_kernel<<<(n + 255) / 256, 256, 0, s>>>(x, m, st, C, lgV);
}

extern "C" void kernel_launch(void* const* d_in, const int* in_sizes, int n_in,
                              void* d_out, int out_size, void* d_ws,
                              size_t ws_size, hipStream_t stream) {
  (void)in_sizes; (void)n_in; (void)out_size; (void)ws_size;
  // setup_inputs() dict order:
  const float* x       = (const float*)d_in[0];
  const float* mask    = (const float*)d_in[1];
  const float* w_enc0  = (const float*)d_in[2];
  const float* w_enc1  = (const float*)d_in[3];
  const float* w_enc2  = (const float*)d_in[4];
  const float* w_enc3  = (const float*)d_in[5];
  const float* w_btd   = (const float*)d_in[6];
  const float* w_btc   = (const float*)d_in[7];
  const float* w_up0   = (const float*)d_in[8];
  const float* w_fuse0 = (const float*)d_in[9];
  const float* w_up1   = (const float*)d_in[10];
  const float* w_fuse1 = (const float*)d_in[11];
  const float* w_up2   = (const float*)d_in[12];
  const float* w_fuse2 = (const float*)d_in[13];
  const float* w_up3   = (const float*)d_in[14];
  const float* w_fuse3 = (const float*)d_in[15];
  const float* w_final = (const float*)d_in[16];
  float* out = (float*)d_out;

  // bump-allocate workspace (floats); all region sizes are multiples of 8
  // floats, so every buffer stays 32B-aligned for b128 stores.
  float* ws = (float*)d_ws;
  size_t off = 0;
  auto alloc = [&](size_t n) { float* p = ws + off; off += n; return p; };
  const int V64 = 64 * 64 * 64, V32 = 32 * 32 * 32, V16 = 16 * 16 * 16;
  const int V8 = 512, V4 = 64, V2 = 8;
  float* m0 = alloc(V64); float* m1 = alloc(V32); float* m2 = alloc(V16);
  float* m3 = alloc(V8);  float* m4 = alloc(V4);  float* m5 = alloc(V2);
  float* hx   = alloc((size_t)8 * V64);
  float* e0   = alloc((size_t)64 * V32);
  float* e1   = alloc((size_t)128 * V16);
  float* e2   = alloc((size_t)256 * V8);
  float* e3   = alloc((size_t)512 * V4);
  float* b0   = alloc((size_t)1024 * V2);
  float* b1   = alloc((size_t)1024 * V2);
  float* cat0 = alloc((size_t)1024 * V4);
  float* f0   = alloc((size_t)512 * V4);
  float* cat1 = alloc((size_t)512 * V8);
  float* f1   = alloc((size_t)256 * V8);
  float* cat2 = alloc((size_t)256 * V16);
  float* f2   = alloc((size_t)128 * V16);
  float* cat3 = alloc((size_t)128 * V32);
  float* f3   = alloc((size_t)64 * V32);
  float* st   = alloc(4096);

  // mask pyramid
  mask_init_kernel<<<(V64 + 255) / 256, 256, 0, stream>>>(mask, m0, V64);
  mask_down_kernel<<<(V32 + 255) / 256, 256, 0, stream>>>(m0, m1, 64, 32);
  mask_down_kernel<<<(V16 + 255) / 256, 256, 0, stream>>>(m1, m2, 32, 16);
  mask_down_kernel<<<(V8 + 255) / 256, 256, 0, stream>>>(m2, m3, 16, 8);
  mask_down_kernel<<<(V4 + 255) / 256, 256, 0, stream>>>(m3, m4, 8, 4);
  mask_down_kernel<<<(V2 + 255) / 256, 256, 0, stream>>>(m4, m5, 4, 2);

  // h = x * m0
  mulmask_kernel<<<(8 * V64 + 255) / 256, 256, 0, stream>>>(x, m0, hx, V64 - 1,
                                                            8 * V64);

  // encoder: conv k3 s2 p1 + BN + ReLU
  run_conv<8, 27, 0, 2, 1, 6, 5>(hx, w_enc0, e0, 64, stream);
  run_bn(e0, m1, st, 64, 15, stream);
  run_conv<64, 27, 0, 2, 1, 5, 4>(e0, w_enc1, e1, 128, stream);
  run_bn(e1, m2, st, 128, 12, stream);
  run_conv<128, 27, 0, 2, 1, 4, 3>(e1, w_enc2, e2, 256, stream);
  run_bn(e2, m3, st, 256, 9, stream);
  run_conv<256, 27, 0, 2, 1, 3, 2>(e2, w_enc3, e3, 512, stream);
  run_bn(e3, m4, st, 512, 6, stream);

  // bottleneck
  run_conv<512, 27, 0, 2, 1, 2, 1>(e3, w_btd, b0, 1024, stream);
  run_bn(b0, m5, st, 1024, 3, stream);
  run_conv<1024, 27, 0, 1, 1, 1, 1>(b0, w_btc, b1, 1024, stream);
  run_bn(b1, m5, st, 1024, 3, stream);

  // decoder stage 0: up(1024->512, 2->4) + concat e3 + 1x1 fuse(1024->512)
  run_conv<1024, 27, 1, 2, 0, 1, 2>(b1, w_up0, cat0, 512, stream);
  run_bn(cat0, m4, st, 512, 6, stream);
  (void)hipMemcpyAsync(cat0 + (size_t)512 * V4, e3,
                       (size_t)512 * V4 * sizeof(float),
                       hipMemcpyDeviceToDevice, stream);
  run_conv<1024, 1, 0, 1, 0, 2, 2>(cat0, w_fuse0, f0, 512, stream);
  run_bn(f0, m4, st, 512, 6, stream);

  // stage 1: up(512->256, 4->8) + concat e2 + fuse(512->256)
  run_conv<512, 27, 1, 2, 0, 2, 3>(f0, w_up1, cat1, 256, stream);
  run_bn(cat1, m3, st, 256, 9, stream);
  (void)hipMemcpyAsync(cat1 + (size_t)256 * V8, e2,
                       (size_t)256 * V8 * sizeof(float),
                       hipMemcpyDeviceToDevice, stream);
  run_conv<512, 1, 0, 1, 0, 3, 3>(cat1, w_fuse1, f1, 256, stream);
  run_bn(f1, m3, st, 256, 9, stream);

  // stage 2: up(256->128, 8->16) + concat e1 + fuse(256->128)
  run_conv<256, 27, 1, 2, 0, 3, 4>(f1, w_up2, cat2, 128, stream);
  run_bn(cat2, m2, st, 128, 12, stream);
  (void)hipMemcpyAsync(cat2 + (size_t)128 * V16, e1,
                       (size_t)128 * V16 * sizeof(float),
                       hipMemcpyDeviceToDevice, stream);
  run_conv<256, 1, 0, 1, 0, 4, 4>(cat2, w_fuse2, f2, 128, stream);
  run_bn(f2, m2, st, 128, 12, stream);

  // stage 3: up(128->64, 16->32) + concat e0 + fuse(128->64)
  run_conv<128, 27, 1, 2, 0, 4, 5>(f2, w_up3, cat3, 64, stream);
  run_bn(cat3, m1, st, 64, 15, stream);
  (void)hipMemcpyAsync(cat3 + (size_t)64 * V32, e0,
                       (size_t)64 * V32 * sizeof(float),
                       hipMemcpyDeviceToDevice, stream);
  run_conv<128, 1, 0, 1, 0, 5, 5>(cat3, w_fuse3, f3, 64, stream);
  run_bn(f3, m1, st, 64, 15, stream);

  // final: up(64->16, 32->64) + BN + ReLU in place on d_out
  run_conv<64, 27, 1, 2, 0, 5, 6>(f3, w_final, out, 16, stream);
  run_bn(out, m0, st, 16, 18, stream);
}
